// RWKV_17111149707624
// MI455X (gfx1250) — matrix operand order repaired and checked by reading, not compiled, not run on hardware
//
#include <hip/hip_runtime.h>
#include <hip/hip_bf16.h>
#include <math.h>

// ---------------- problem constants (match reference) ----------------
#define HID   1024
#define NATTN 1024
#define NHEAD 16
#define CTX   1024
#define HS    2560            // 5*NFFN/2
#define HEAD  64              // NATTN / NHEAD
#define BSZ   8
#define TLEN  1024
#define MROWS (BSZ*TLEN)      // 8192

typedef __bf16 bf16;
typedef __attribute__((ext_vector_type(16))) __bf16 v16bf;
typedef __attribute__((ext_vector_type(8)))  __bf16 v8bf;
typedef __attribute__((ext_vector_type(8)))  float  v8f;

// ---------------- fragment loaders (CDNA5 wave32 WMMA layouts) ----------------
// A: 16x32 bf16, row-major source with leading dim lda.
// lanes 0-15 : row m, K = k0+0..15
// lanes 16-31: row m, K = k0+16..31
__device__ __forceinline__ v16bf load_a_frag(const bf16* __restrict__ A, int lda,
                                             int m0, int k0, int lane) {
    const int half = lane >> 4;
    const int m    = m0 + (lane & 15);
    const bf16* p  = A + (size_t)m * lda + k0 + half * 16;
    v8bf lo = *(const v8bf*)(p);
    v8bf hi = *(const v8bf*)(p + 8);
    v16bf a;
#pragma unroll
    for (int i = 0; i < 8; ++i) { a[i] = lo[i]; a[i + 8] = hi[i]; }
    return a;
}

// B: 32x16 bf16 pre-packed per lane: dst[((kb*ntiles+nt)*32+lane)*16+e]
// holds B[k= kb*32 + (lane>=16)*16 + e][n = nt*16 + (lane&15)]
__device__ __forceinline__ v16bf load_b_frag(const bf16* __restrict__ Bp, int ntiles,
                                             int kb, int nt, int lane) {
    const bf16* p = Bp + (((size_t)kb * ntiles + nt) * 32 + lane) * 16;
    return *(const v16bf*)p;
}

__device__ __forceinline__ float sigf(float x) { return 1.f / (1.f + expf(-x)); }

// ---------------- pack f32 weight [N(out)][K(in)] -> bf16 B-fragments ----------------
__global__ void pack_weight(const float* __restrict__ W, bf16* __restrict__ Bp,
                            int N, int K) {
    size_t idx = (size_t)blockIdx.x * blockDim.x + threadIdx.x;
    size_t total = (size_t)N * K;
    if (idx >= total) return;
    int e    = idx & 15;
    int lane = (idx >> 4) & 31;
    size_t rem = idx >> 9;
    int ntiles = N >> 4;
    int nt = (int)(rem % ntiles);
    int kb = (int)(rem / ntiles);
    int k  = kb * 32 + ((lane >> 4) << 4) + e;
    int n  = nt * 16 + (lane & 15);
    Bp[idx] = (bf16)W[(size_t)n * K + k];
}

// ---------------- build causal time-mix matrix Wh[h][t][u] in bf16 ----------------
// w[h,t,u] = (u<=t) ? time_w[h, u-t+CTX-1] * alpha[h,u] * beta[h,t] : 0
__global__ void build_wh(const float* __restrict__ time_w, const float* __restrict__ alpha,
                         const float* __restrict__ beta, bf16* __restrict__ Wh) {
    size_t idx = (size_t)blockIdx.x * blockDim.x + threadIdx.x;
    size_t total = (size_t)NHEAD * CTX * CTX;
    if (idx >= total) return;
    int u = idx & (CTX - 1);
    int t = (int)((idx >> 10) & (CTX - 1));
    int h = (int)(idx >> 20);
    float v = 0.f;
    if (u <= t) {
        v = time_w[h * CTX + (u - t + CTX - 1)] * alpha[h * CTX + u] * beta[h * CTX + t];
    }
    Wh[idx] = (bf16)v;
}

// ---------------- pack kv f32 [B*T][NATTN] -> per-(b,h) B-fragments ----------------
// layout: [bh(7b)][kb(5b)][nt(2b)][lane(5b)][e(4b)]
__global__ void pack_kv(const float* __restrict__ kvf, bf16* __restrict__ Bp) {
    size_t idx = (size_t)blockIdx.x * blockDim.x + threadIdx.x;
    size_t total = (size_t)BSZ * NHEAD * TLEN * HEAD;
    if (idx >= total) return;
    int e    = idx & 15;
    int lane = (int)((idx >> 4) & 31);
    int nt   = (int)((idx >> 9) & 3);
    int kb   = (int)((idx >> 11) & 31);
    int bh   = (int)(idx >> 16);
    int b = bh >> 4, h = bh & 15;
    int k = kb * 32 + ((lane >> 4) << 4) + e;
    int n = nt * 16 + (lane & 15);
    Bp[idx] = (bf16)kvf[((size_t)b * TLEN + k) * NATTN + h * HEAD + n];
}

// ---------------- generic bf16 WMMA GEMM: C[M][N] = A[M][K] * Bpacked + bias ----------
// block = 128 threads (4 waves); wave -> 32x64 tile (2 M x 4 N WMMA tiles).
// Software pipelined with last-iteration peel (no conditional selects); launch_bounds
// releases the occupancy register cap so acc + both pipeline stages stay in VGPRs.
// grid(M/32, ceil(N/256))
__global__ __launch_bounds__(128, 1)
void gemm_bf16_wmma(const bf16* __restrict__ A, const bf16* __restrict__ Bp,
                    const float* __restrict__ bias, float* __restrict__ C,
                    int M, int N, int K) {
    const int lane = threadIdx.x & 31;
    const int wave = threadIdx.x >> 5;
    const int m0   = blockIdx.x * 32;
    const int col0 = blockIdx.y * 256 + wave * 64;
    if (col0 >= N) return;                 // wave-uniform: EXEC stays all-ones for WMMA
    const int ntiles = N >> 4;
    const int nt0    = col0 >> 4;
    const int nkb    = K >> 5;

    v8f acc[2][4] = {};
    v16bf a[2], b[4];
    a[0] = load_a_frag(A, K, m0,      0, lane);
    a[1] = load_a_frag(A, K, m0 + 16, 0, lane);
#pragma unroll
    for (int i = 0; i < 4; ++i) b[i] = load_b_frag(Bp, ntiles, 0, nt0 + i, lane);

#pragma unroll 2
    for (int kb = 0; kb < nkb - 1; ++kb) {
        const int k1 = (kb + 1) * 32;
        v16bf an[2], bn[4];
        an[0] = load_a_frag(A, K, m0,      k1, lane);
        an[1] = load_a_frag(A, K, m0 + 16, k1, lane);
#pragma unroll
        for (int i = 0; i < 4; ++i) bn[i] = load_b_frag(Bp, ntiles, kb + 1, nt0 + i, lane);
#pragma unroll
        for (int mi = 0; mi < 2; ++mi)
#pragma unroll
            for (int i = 0; i < 4; ++i)
                acc[mi][i] = __builtin_amdgcn_wmma_f32_16x16x32_bf16(
                    false, a[mi], false, b[i], (short)0, acc[mi][i], false, false);
        a[0] = an[0]; a[1] = an[1];
#pragma unroll
        for (int i = 0; i < 4; ++i) b[i] = bn[i];
    }
    // peeled final K-step (no prefetch)
#pragma unroll
    for (int mi = 0; mi < 2; ++mi)
#pragma unroll
        for (int i = 0; i < 4; ++i)
            acc[mi][i] = __builtin_amdgcn_wmma_f32_16x16x32_bf16(
                false, a[mi], false, b[i], (short)0, acc[mi][i], false, false);

    const int half = lane >> 4, n = lane & 15;
#pragma unroll
    for (int mi = 0; mi < 2; ++mi)
#pragma unroll
        for (int i = 0; i < 4; ++i) {
            int col = col0 + i * 16 + n;
            float bv = bias ? bias[col] : 0.f;
#pragma unroll
            for (int j = 0; j < 8; ++j) {
                int m = m0 + mi * 16 + j + 8 * half;
                C[(size_t)m * N + col] = acc[mi][i][j] + bv;
            }
        }
}

// ---------------- causal WKV GEMM: Out[b,t,h*64+c] = sum_{u<=t} Wh[h][t][u]*kv[b,u,h,c] ----
// block = 128 (4 waves, each a 32x64 tile); grid(T/128, 1, B*H); pipelined as above.
__global__ __launch_bounds__(128, 1)
void wkv_gemm_wmma(const bf16* __restrict__ Wh, const bf16* __restrict__ KVp,
                   float* __restrict__ Out) {
    const int lane = threadIdx.x & 31;
    const int wave = threadIdx.x >> 5;
    const int bh = blockIdx.z;
    const int b = bh >> 4, h = bh & 15;
    const int m0 = (blockIdx.x * 4 + wave) * 32;

    const bf16* A  = Wh  + (size_t)h  * CTX * CTX;
    const bf16* Bp = KVp + (size_t)bh * (TLEN / 32) * 4 * 32 * 16;

    v8f acc[2][4] = {};
    const int nkb = ((m0 + 31) >> 5) + 1;    // causal: only blocks with u <= t
    v16bf a[2], bfr[4];
    a[0] = load_a_frag(A, CTX, m0,      0, lane);
    a[1] = load_a_frag(A, CTX, m0 + 16, 0, lane);
#pragma unroll
    for (int i = 0; i < 4; ++i) bfr[i] = load_b_frag(Bp, 4, 0, i, lane);

#pragma unroll 2
    for (int kb = 0; kb < nkb - 1; ++kb) {
        const int k1 = (kb + 1) * 32;
        v16bf an[2], bn[4];
        an[0] = load_a_frag(A, CTX, m0,      k1, lane);
        an[1] = load_a_frag(A, CTX, m0 + 16, k1, lane);
#pragma unroll
        for (int i = 0; i < 4; ++i) bn[i] = load_b_frag(Bp, 4, kb + 1, i, lane);
#pragma unroll
        for (int mi = 0; mi < 2; ++mi)
#pragma unroll
            for (int i = 0; i < 4; ++i)
                acc[mi][i] = __builtin_amdgcn_wmma_f32_16x16x32_bf16(
                    false, a[mi], false, bfr[i], (short)0, acc[mi][i], false, false);
        a[0] = an[0]; a[1] = an[1];
#pragma unroll
        for (int i = 0; i < 4; ++i) bfr[i] = bn[i];
    }
#pragma unroll
    for (int mi = 0; mi < 2; ++mi)
#pragma unroll
        for (int i = 0; i < 4; ++i)
            acc[mi][i] = __builtin_amdgcn_wmma_f32_16x16x32_bf16(
                false, a[mi], false, bfr[i], (short)0, acc[mi][i], false, false);

    const int half = lane >> 4, n = lane & 15;
#pragma unroll
    for (int mi = 0; mi < 2; ++mi)
#pragma unroll
        for (int i = 0; i < 4; ++i)
#pragma unroll
            for (int j = 0; j < 8; ++j) {
                int m = m0 + mi * 16 + j + 8 * half;
                Out[((size_t)b * TLEN + m) * NATTN + h * HEAD + i * 16 + n] = acc[mi][i][j];
            }
}

// ---------------- elementwise kernels ----------------
__global__ void ln_rows(const float* __restrict__ X, const float* __restrict__ g,
                        const float* __restrict__ b, float* __restrict__ Y, int C) {
    __shared__ float s1[256], s2[256];
    const int row = blockIdx.x;
    const float* x = X + (size_t)row * C;
    float sum = 0.f, sq = 0.f;
    for (int c = threadIdx.x; c < C; c += blockDim.x) { float v = x[c]; sum += v; sq += v * v; }
    s1[threadIdx.x] = sum; s2[threadIdx.x] = sq;
    __syncthreads();
    for (int off = blockDim.x >> 1; off > 0; off >>= 1) {
        if ((int)threadIdx.x < off) { s1[threadIdx.x] += s1[threadIdx.x + off];
                                      s2[threadIdx.x] += s2[threadIdx.x + off]; }
        __syncthreads();
    }
    const float mean = s1[0] / C;
    const float var  = s2[0] / C - mean * mean;
    const float inv  = rsqrtf(var + 1e-5f);
    float* y = Y + (size_t)row * C;
    for (int c = threadIdx.x; c < C; c += blockDim.x)
        y[c] = (x[c] - mean) * inv * g[c] + b[c];
}

// time-shift (first C/2 channels delayed one step) + f32 -> bf16
__global__ void shift_to_bf16(const float* __restrict__ Xn, bf16* __restrict__ Xs, int C) {
    size_t i = (size_t)blockIdx.x * blockDim.x + threadIdx.x;
    size_t total = (size_t)MROWS * C;
    if (i >= total) return;
    size_t row = i / C;
    int c = (int)(i - row * C);
    int t = (int)(row & (TLEN - 1));
    float v;
    if (c < C / 2) v = (t > 0) ? Xn[(row - 1) * C + c] : 0.f;
    else           v = Xn[i];
    Xs[i] = (bf16)v;
}

__global__ void expkv_kernel(float* __restrict__ k, const float* __restrict__ v,
                             float* __restrict__ kv, size_t n) {
    size_t i = (size_t)blockIdx.x * blockDim.x + threadIdx.x;
    if (i >= n) return;
    float ke = expf(fminf(fmaxf(k[i], -60.f), 30.f));
    k[i] = ke;
    kv[i] = ke * v[i];
}

__global__ void cumsum_time(const float* __restrict__ k, float* __restrict__ s, int C) {
    int i = blockIdx.x * blockDim.x + threadIdx.x;   // over B*C columns
    if (i >= BSZ * C) return;
    int b = i / C, c = i - b * C;
    const float* kp = k + (size_t)b * TLEN * C + c;
    float*       sp = s + (size_t)b * TLEN * C + c;
    float acc = 0.f;
    for (int t = 0; t < TLEN; ++t) { acc += kp[(size_t)t * C]; sp[(size_t)t * C] = acc; }
}

__global__ void rwkv_gate(const float* __restrict__ r, const float* __restrict__ wkv,
                          const float* __restrict__ sumk, bf16* __restrict__ out, size_t n) {
    size_t i = (size_t)blockIdx.x * blockDim.x + threadIdx.x;
    if (i >= n) return;
    out[i] = (bf16)(sigf(r[i]) * wkv[i] / sumk[i]);
}

__global__ void att_resid(const float* __restrict__ x, const float* __restrict__ att,
                          const float* __restrict__ gamma, float* __restrict__ x2) {
    size_t i = (size_t)blockIdx.x * blockDim.x + threadIdx.x;
    size_t total = (size_t)MROWS * HID;
    if (i >= total) return;
    int t = (int)((i / HID) & (TLEN - 1));
    x2[i] = x[i] + att[i] * gamma[t];
}

__global__ void mish_mul(const float* __restrict__ k2, const float* __restrict__ v2,
                         bf16* __restrict__ h, size_t n) {
    size_t i = (size_t)blockIdx.x * blockDim.x + threadIdx.x;
    if (i >= n) return;
    float kk = k2[i];
    float sp = (kk > 20.f) ? kk : log1pf(expf(kk));   // softplus (stable)
    h[i] = (bf16)(kk * tanhf(sp) * v2[i]);
}

__global__ void final_out(const float* __restrict__ x2, const float* __restrict__ r2,
                          const float* __restrict__ wkv2, float* __restrict__ out, size_t n) {
    size_t i = (size_t)blockIdx.x * blockDim.x + threadIdx.x;
    if (i >= n) return;
    out[i] = x2[i] + sigf(r2[i]) * wkv2[i];
}

// ---------------- host orchestration ----------------
static inline int ceil_div_i(size_t a, int b) { return (int)((a + b - 1) / b); }

extern "C" void kernel_launch(void* const* d_in, const int* in_sizes, int n_in,
                              void* d_out, int out_size, void* d_ws, size_t ws_size,
                              hipStream_t stream) {
    (void)in_sizes; (void)n_in; (void)out_size; (void)ws_size;

    const float* x        = (const float*)d_in[0];
    const float* ln1_g    = (const float*)d_in[1];
    const float* ln1_b    = (const float*)d_in[2];
    const float* ln2_g    = (const float*)d_in[3];
    const float* ln2_b    = (const float*)d_in[4];
    const float* time_w   = (const float*)d_in[5];
    const float* time_al  = (const float*)d_in[6];
    const float* time_be  = (const float*)d_in[7];
    const float* time_ga  = (const float*)d_in[8];
    const float* att_k_w  = (const float*)d_in[9];
    const float* att_k_b  = (const float*)d_in[10];
    const float* att_v_w  = (const float*)d_in[11];
    const float* att_v_b  = (const float*)d_in[12];
    const float* att_r_w  = (const float*)d_in[13];
    const float* att_r_b  = (const float*)d_in[14];
    const float* att_o_w  = (const float*)d_in[15];
    const float* att_o_b  = (const float*)d_in[16];
    const float* mlp_k_w  = (const float*)d_in[17];
    const float* mlp_k_b  = (const float*)d_in[18];
    const float* mlp_v_w  = (const float*)d_in[19];
    const float* mlp_v_b  = (const float*)d_in[20];
    const float* mlp_r_w  = (const float*)d_in[21];
    const float* mlp_r_b  = (const float*)d_in[22];
    const float* mlp_w_w  = (const float*)d_in[23];
    const float* mlp_w_b  = (const float*)d_in[24];
    float* out = (float*)d_out;

    size_t off = 0;
    auto alloc = [&](size_t bytes) -> void* {
        void* p = (char*)d_ws + off;
        off += (bytes + 255) & ~(size_t)255;
        return p;
    };

    // ---- persistent region (lives across both phases) ----
    bf16* wp_att_k = (bf16*)alloc((size_t)HID * NATTN * 2);
    bf16* wp_att_v = (bf16*)alloc((size_t)HID * NATTN * 2);
    bf16* wp_att_r = (bf16*)alloc((size_t)HID * NATTN * 2);
    bf16* wp_att_o = (bf16*)alloc((size_t)NATTN * HID * 2);
    bf16* wp_mlp_k = (bf16*)alloc((size_t)HID * HS * 2);
    bf16* wp_mlp_v = (bf16*)alloc((size_t)HID * HS * 2);
    bf16* wp_mlp_r = (bf16*)alloc((size_t)HID * HID * 2);
    bf16* wp_mlp_w = (bf16*)alloc((size_t)HS * HID * 2);
    bf16* Wh       = (bf16*)alloc((size_t)NHEAD * CTX * CTX * 2);
    float* xn      = (float*)alloc((size_t)MROWS * HID * 4);   // reused for LN2
    bf16*  xs      = (bf16*) alloc((size_t)MROWS * HID * 2);   // reused for shift2
    float* x2      = (float*)alloc((size_t)MROWS * HID * 4);
    const size_t checkpoint = off;                              // phase scratch reuse point

    // ---- phase-1 scratch ----
    float* kf   = (float*)alloc((size_t)MROWS * NATTN * 4);
    float* vf   = (float*)alloc((size_t)MROWS * NATTN * 4);
    float* rf   = (float*)alloc((size_t)MROWS * NATTN * 4);
    float* kvf  = (float*)alloc((size_t)MROWS * NATTN * 4);
    float* sumk = (float*)alloc((size_t)MROWS * NATTN * 4);
    bf16*  kvp  = (bf16*) alloc((size_t)BSZ * NHEAD * TLEN * HEAD * 2);
    float* wkv  = (float*)alloc((size_t)MROWS * NATTN * 4);
    bf16*  rA   = (bf16*) alloc((size_t)MROWS * NATTN * 2);
    float* att  = (float*)alloc((size_t)MROWS * HID * 4);

    const int EB = 256;

    // ---- weight packing + Wh build ----
    pack_weight<<<ceil_div_i((size_t)NATTN * HID, EB), EB, 0, stream>>>(att_k_w, wp_att_k, NATTN, HID);
    pack_weight<<<ceil_div_i((size_t)NATTN * HID, EB), EB, 0, stream>>>(att_v_w, wp_att_v, NATTN, HID);
    pack_weight<<<ceil_div_i((size_t)NATTN * HID, EB), EB, 0, stream>>>(att_r_w, wp_att_r, NATTN, HID);
    pack_weight<<<ceil_div_i((size_t)HID * NATTN, EB), EB, 0, stream>>>(att_o_w, wp_att_o, HID, NATTN);
    pack_weight<<<ceil_div_i((size_t)HS * HID, EB), EB, 0, stream>>>(mlp_k_w, wp_mlp_k, HS, HID);
    pack_weight<<<ceil_div_i((size_t)HS * HID, EB), EB, 0, stream>>>(mlp_v_w, wp_mlp_v, HS, HID);
    pack_weight<<<ceil_div_i((size_t)HID * HID, EB), EB, 0, stream>>>(mlp_r_w, wp_mlp_r, HID, HID);
    pack_weight<<<ceil_div_i((size_t)HID * HS, EB), EB, 0, stream>>>(mlp_w_w, wp_mlp_w, HID, HS);
    build_wh<<<ceil_div_i((size_t)NHEAD * CTX * CTX, EB), EB, 0, stream>>>(time_w, time_al, time_be, Wh);

    // ---- phase 1: time-mix (WKV attention) ----
    ln_rows<<<MROWS, 256, 0, stream>>>(x, ln1_g, ln1_b, xn, HID);
    shift_to_bf16<<<ceil_div_i((size_t)MROWS * HID, EB), EB, 0, stream>>>(xn, xs, HID);

    dim3 gemm_blk(128);
    dim3 g1024(MROWS / 32, NATTN / 256);
    gemm_bf16_wmma<<<g1024, gemm_blk, 0, stream>>>(xs, wp_att_k, att_k_b, kf, MROWS, NATTN, HID);
    gemm_bf16_wmma<<<g1024, gemm_blk, 0, stream>>>(xs, wp_att_v, att_v_b, vf, MROWS, NATTN, HID);
    gemm_bf16_wmma<<<g1024, gemm_blk, 0, stream>>>(xs, wp_att_r, att_r_b, rf, MROWS, NATTN, HID);

    expkv_kernel<<<ceil_div_i((size_t)MROWS * NATTN, EB), EB, 0, stream>>>(kf, vf, kvf, (size_t)MROWS * NATTN);
    cumsum_time<<<ceil_div_i((size_t)BSZ * NATTN, EB), EB, 0, stream>>>(kf, sumk, NATTN);
    pack_kv<<<ceil_div_i((size_t)BSZ * NHEAD * TLEN * HEAD, EB), EB, 0, stream>>>(kvf, kvp);

    dim3 gwkv(TLEN / 128, 1, BSZ * NHEAD);
    wkv_gemm_wmma<<<gwkv, gemm_blk, 0, stream>>>(Wh, kvp, wkv);

    rwkv_gate<<<ceil_div_i((size_t)MROWS * NATTN, EB), EB, 0, stream>>>(rf, wkv, sumk, rA, (size_t)MROWS * NATTN);
    gemm_bf16_wmma<<<g1024, gemm_blk, 0, stream>>>(rA, wp_att_o, att_o_b, att, MROWS, HID, NATTN);
    att_resid<<<ceil_div_i((size_t)MROWS * HID, EB), EB, 0, stream>>>(x, att, time_ga, x2);

    // ---- phase 2: channel-mix (reuse phase-1 scratch region) ----
    off = checkpoint;
    float* k2   = (float*)alloc((size_t)MROWS * HS * 4);
    float* v2   = (float*)alloc((size_t)MROWS * HS * 4);
    float* r2   = (float*)alloc((size_t)MROWS * HID * 4);
    bf16*  hb   = (bf16*) alloc((size_t)MROWS * HS * 2);
    float* wkv2 = (float*)alloc((size_t)MROWS * HID * 4);

    ln_rows<<<MROWS, 256, 0, stream>>>(x2, ln2_g, ln2_b, xn, HID);
    shift_to_bf16<<<ceil_div_i((size_t)MROWS * HID, EB), EB, 0, stream>>>(xn, xs, HID);

    dim3 g2560(MROWS / 32, (HS + 255) / 256);
    gemm_bf16_wmma<<<g2560, gemm_blk, 0, stream>>>(xs, wp_mlp_k, mlp_k_b, k2, MROWS, HS, HID);
    gemm_bf16_wmma<<<g2560, gemm_blk, 0, stream>>>(xs, wp_mlp_v, mlp_v_b, v2, MROWS, HS, HID);
    gemm_bf16_wmma<<<g1024, gemm_blk, 0, stream>>>(xs, wp_mlp_r, mlp_r_b, r2, MROWS, HID, HID);

    mish_mul<<<ceil_div_i((size_t)MROWS * HS, EB), EB, 0, stream>>>(k2, v2, hb, (size_t)MROWS * HS);
    gemm_bf16_wmma<<<g1024, gemm_blk, 0, stream>>>(hb, wp_mlp_w, mlp_w_b, wkv2, MROWS, HID, HS);
    final_out<<<ceil_div_i((size_t)MROWS * HID, EB), EB, 0, stream>>>(x2, r2, wkv2, out, (size_t)MROWS * HID);
}
